// Actor_57260503990563
// MI455X (gfx1250) — matrix operand order repaired and checked by reading, not compiled, not run on hardware
//
#include <hip/hip_runtime.h>
#include <hip/hip_bf16.h>

typedef __attribute__((ext_vector_type(16))) _Float16 v16h;
typedef __attribute__((ext_vector_type(8)))  _Float16 v8h;
typedef __attribute__((ext_vector_type(8)))  float    v8f;
typedef __attribute__((ext_vector_type(4)))  float    v4f;

#define NTHREADS 256

// LDS layout (element counts). Strides padded to kill bank conflicts:
//  W0T: [1024 u][40 k]  (K=20 padded to 32, stride 40)  -> 40960 halves
//  W1T: [16*64 u][72 k] (K=64, stride 72)               -> 73728 halves
//  W2T: [64 u][72 k]                                    ->  4608 halves
//  ACTA/ACTB: [16 rows][1032 units]                     -> 16512 halves each
//  W3: [16][64] f32; B0,B1: [1024] f32; B2: [64]; B3: [16]
#define W0T_OFF   0
#define W1T_OFF   (W0T_OFF + 1024*40)
#define W2T_OFF   (W1T_OFF + 1024*72)
#define ACTA_OFF  (W2T_OFF + 64*72)
#define ACTB_OFF  (ACTA_OFF + 16*1032)
#define HALVES_END (ACTB_OFF + 16*1032)
#define SMEM_BYTES (HALVES_END*2 + (1024*3 + 64 + 16)*4)

__device__ __forceinline__ v16h cat8(v8h lo, v8h hi) {
  union { v16h v; v8h h[2]; } u;
  u.h[0] = lo; u.h[1] = hi;
  return u.v;
}

__device__ __forceinline__ v8f cat4f(v4f a, v4f b) {
  v8f r;
  #pragma unroll
  for (int i = 0; i < 4; i++) { r[i] = a[i]; r[4+i] = b[i]; }
  return r;
}

__device__ __forceinline__ v8f wmma16(v16h a, v16h b, v8f c) {
  return __builtin_amdgcn_wmma_f32_16x16x32_f16(false, a, false, b,
                                                (short)0, c, false, false);
}

// 8 back-to-back gfx1250 TRANS tanh ops; consecutive independent trans ops
// cover each other's co-exec hazard, single trailing v_nop covers the last.
// Not volatile: no side effects, lets the scheduler overlap it with loads.
__device__ __forceinline__ v8f tanh8(v8f v) {
  float y0, y1, y2, y3, y4, y5, y6, y7;
  asm("v_tanh_f32 %0, %8\n\t"
      "v_tanh_f32 %1, %9\n\t"
      "v_tanh_f32 %2, %10\n\t"
      "v_tanh_f32 %3, %11\n\t"
      "v_tanh_f32 %4, %12\n\t"
      "v_tanh_f32 %5, %13\n\t"
      "v_tanh_f32 %6, %14\n\t"
      "v_tanh_f32 %7, %15\n\t"
      "v_nop"
      : "=&v"(y0), "=&v"(y1), "=&v"(y2), "=&v"(y3),
        "=&v"(y4), "=&v"(y5), "=&v"(y6), "=&v"(y7)
      : "v"(v[0]), "v"(v[1]), "v"(v[2]), "v"(v[3]),
        "v"(v[4]), "v"(v[5]), "v"(v[6]), "v"(v[7]));
  return (v8f){y0, y1, y2, y3, y4, y5, y6, y7};
}

// tanh + f16 pack + one ds_store_b128 (8 consecutive units, one batch row)
__device__ __forceinline__ void act_store(_Float16* dst, v8f acc) {
  v8f t = tanh8(acc);
  v8h p;
  #pragma unroll
  for (int i = 0; i < 8; i++) p[i] = (_Float16)t[i];
  *(v8h*)dst = p;
}

__global__ __launch_bounds__(NTHREADS, 1)
void actor_fused_kernel(const float* __restrict__ x,
                        const float* __restrict__ dg_w0, const float* __restrict__ dg_b0,
                        const float* __restrict__ dg_w1, const float* __restrict__ dg_b1,
                        const float* __restrict__ dg_w2, const float* __restrict__ dg_b2,
                        const float* __restrict__ dg_w3, const float* __restrict__ dg_b3,
                        const float* __restrict__ ev_w0, const float* __restrict__ ev_b0,
                        const float* __restrict__ ev_w1, const float* __restrict__ ev_b1,
                        const float* __restrict__ ev_w2, const float* __restrict__ ev_b2,
                        const float* __restrict__ ac_w0, const float* __restrict__ ac_b0,
                        const float* __restrict__ ac_w1, const float* __restrict__ ac_b1,
                        const float* __restrict__ ac_w2, const float* __restrict__ ac_b2,
                        float* __restrict__ out, int Bsz)
{
  extern __shared__ char smem_raw[];
  _Float16* W0T  = (_Float16*)smem_raw + W0T_OFF;
  _Float16* W1T  = (_Float16*)smem_raw + W1T_OFF;
  _Float16* W2T  = (_Float16*)smem_raw + W2T_OFF;
  _Float16* ACTA = (_Float16*)smem_raw + ACTA_OFF;
  _Float16* ACTB = (_Float16*)smem_raw + ACTB_OFF;
  float* W3 = (float*)(smem_raw + HALVES_END*2);
  float* B0 = W3 + 1024;
  float* B1 = B0 + 1024;
  float* B2 = B1 + 1024;
  float* B3 = B2 + 64;

  const int tid = threadIdx.x;

  // ---------------- Build f16 weight images in LDS (once per block) --------
  // [unit][k] layout, exactly what A-fragments (weights-as-A) want.
  for (int i = tid; i < 1024*40; i += NTHREADS) {
    int n = i / 40, k = i % 40;
    float v = 0.f;
    if (k < 32) {
      if (n < 64) {                       // DG
        if (k < 20) v = dg_w0[n*20 + k];
      } else if (n < 384) {               // EV i: inputs (x0, x[5+i])
        int br = (n - 64) >> 6, j = (n - 64) & 63;
        if (k == 0)            v = ev_w0[(br*64 + j)*2 + 0];
        else if (k == 5 + br)  v = ev_w0[(br*64 + j)*2 + 1];
      } else {                            // AC i: inputs (x0, x4, x[10+i])
        int br = (n - 384) >> 6, j = (n - 384) & 63;
        if (k == 0)            v = ac_w0[(br*64 + j)*3 + 0];
        else if (k == 4)       v = ac_w0[(br*64 + j)*3 + 1];
        else if (k == 10 + br) v = ac_w0[(br*64 + j)*3 + 2];
      }
    }
    W0T[i] = (_Float16)v;
  }
  for (int i = tid; i < 1024*72; i += NTHREADS) {
    int n = i / 72, k = i % 72;
    float v = 0.f;
    if (k < 64) {
      int net = n >> 6, j = n & 63;
      if (net == 0)      v = dg_w1[j*64 + k];
      else if (net < 6)  v = ev_w1[((net-1)*64 + j)*64 + k];
      else               v = ac_w1[((net-6)*64 + j)*64 + k];
    }
    W1T[i] = (_Float16)v;
  }
  for (int i = tid; i < 64*72; i += NTHREADS) {
    int j = i / 72, k = i % 72;
    W2T[i] = (_Float16)((k < 64) ? dg_w2[j*64 + k] : 0.f);
  }
  for (int i = tid; i < 1024; i += NTHREADS) {
    int o = i >> 6, k = i & 63;
    W3[i] = (o == 0) ? dg_w3[k] : (o < 6) ? ev_w2[(o-1)*64 + k] : ac_w2[(o-6)*64 + k];
    B0[i] = (i < 64) ? dg_b0[i] : (i < 384) ? ev_b0[i-64] : ac_b0[i-384];
    B1[i] = (i < 64) ? dg_b1[i] : (i < 384) ? ev_b1[i-64] : ac_b1[i-384];
  }
  if (tid < 64) B2[tid] = dg_b2[tid];
  if (tid < 16) B3[tid] = (tid == 0) ? dg_b3[0] : (tid < 6) ? ev_b2[tid-1] : ac_b2[tid-6];
  __syncthreads();

  const int wave = tid >> 5;
  const int lane = tid & 31;
  const int l16  = lane & 15;   // batch row within tile (N lane)
  const int hi8  = (lane >> 4) << 3;
  const int hi16 = (lane >> 4) << 4;
  const int ntiles = Bsz >> 4;

  // -------- software-pipelined x fetch: raw f32 regs carried across iters --
  v4f p0 = {0.f,0.f,0.f,0.f}, p1 = p0, p2 = p0, p3 = p0;
  {
    int T0 = blockIdx.x;
    if (T0 >= ntiles) T0 = ntiles - 1;
    int row = (T0 << 4) + l16;
    if (row >= Bsz) row = Bsz - 1;
    const float* xr = x + (size_t)row * 20;
    if (hi8 == 0) {
      p0 = *(const v4f*)(xr + 0);  p1 = *(const v4f*)(xr + 4);
      p2 = *(const v4f*)(xr + 8);  p3 = *(const v4f*)(xr + 12);
    } else {
      p0 = *(const v4f*)(xr + 16);
    }
  }

  for (int T = blockIdx.x; T < ntiles; T += gridDim.x) {
    const int rowbase = T << 4;

    // ---------------- Stage A: D[u,b] = W0[u,k] x X[k,b] ------------------
    {
      // B-frag (x): lane l holds batch col l16, K = hi16 + (0..15), pad >=20
      v16h bx;
      if (hi8 == 0) {
        #pragma unroll
        for (int i = 0; i < 4; i++) {
          bx[i]      = (_Float16)p0[i];
          bx[4 + i]  = (_Float16)p1[i];
          bx[8 + i]  = (_Float16)p2[i];
          bx[12 + i] = (_Float16)p3[i];
        }
      } else {
        #pragma unroll
        for (int i = 0; i < 4; i++) {
          bx[i]      = (_Float16)p0[i];
          bx[4 + i]  = (_Float16)0.f;
          bx[8 + i]  = (_Float16)0.f;
          bx[12 + i] = (_Float16)0.f;
        }
      }
      #pragma unroll
      for (int t = 0; t < 8; t++) {
        const int u16 = ((wave << 3) + t) << 4;   // 16-unit M tile
        const _Float16* wrow = &W0T[(u16 + l16)*40];
        v16h a = cat8(*(const v8h*)&wrow[hi16], *(const v8h*)&wrow[hi16 + 8]);
        v8f acc = cat4f(*(const v4f*)&B0[u16 + hi8],
                        *(const v4f*)&B0[u16 + hi8 + 4]);
        acc = wmma16(a, bx, acc);
        act_store(&ACTA[l16*1032 + u16 + hi8], acc);
      }
    }

    // Issue next tile's x loads now; they complete during stages B/C/D.
    {
      int Tn = T + (int)gridDim.x;
      if (Tn >= ntiles) Tn = T;
      int row = (Tn << 4) + l16;
      if (row >= Bsz) row = Bsz - 1;
      const float* xr = x + (size_t)row * 20;
      if (hi8 == 0) {
        p0 = *(const v4f*)(xr + 0);  p1 = *(const v4f*)(xr + 4);
        p2 = *(const v4f*)(xr + 8);  p3 = *(const v4f*)(xr + 12);
      } else {
        p0 = *(const v4f*)(xr + 16);
      }
    }
    __syncthreads();

    // ---------------- Stage B: block-diagonal 16 x (64->64) ---------------
    #pragma unroll
    for (int nn = 0; nn < 2; nn++) {
      const int net = (wave << 1) + nn;
      const int kb = net*64 + hi16;
      // B-frags (activations): K runs of 16 consecutive units
      v16h b0 = cat8(*(const v8h*)&ACTA[l16*1032 + kb],
                     *(const v8h*)&ACTA[l16*1032 + kb + 8]);
      v16h b1 = cat8(*(const v8h*)&ACTA[l16*1032 + kb + 32],
                     *(const v8h*)&ACTA[l16*1032 + kb + 40]);
      #pragma unroll
      for (int t = 0; t < 4; t++) {
        const int ub = net*64 + t*16;
        const _Float16* wrow = &W1T[(ub + l16)*72];
        v16h a0 = cat8(*(const v8h*)&wrow[hi16],      *(const v8h*)&wrow[hi16 + 8]);
        v16h a1 = cat8(*(const v8h*)&wrow[32 + hi16], *(const v8h*)&wrow[40 + hi16]);
        v8f acc = cat4f(*(const v4f*)&B1[ub + hi8],
                        *(const v4f*)&B1[ub + hi8 + 4]);
        acc = wmma16(a0, b0, acc);
        acc = wmma16(a1, b1, acc);
        act_store(&ACTB[l16*1032 + ub + hi8], acc);
      }
    }
    __syncthreads();

    // ---------------- Stage C: DG extra 64->64 (waves 0..3) ---------------
    if (wave < 4) {
      v16h b0 = cat8(*(const v8h*)&ACTB[l16*1032 + hi16],
                     *(const v8h*)&ACTB[l16*1032 + hi16 + 8]);
      v16h b1 = cat8(*(const v8h*)&ACTB[l16*1032 + hi16 + 32],
                     *(const v8h*)&ACTB[l16*1032 + hi16 + 40]);
      const int ub = wave << 4;
      const _Float16* wrow = &W2T[(ub + l16)*72];
      v16h a0 = cat8(*(const v8h*)&wrow[hi16],      *(const v8h*)&wrow[hi16 + 8]);
      v16h a1 = cat8(*(const v8h*)&wrow[32 + hi16], *(const v8h*)&wrow[40 + hi16]);
      v8f acc = cat4f(*(const v4f*)&B2[ub + hi8],
                      *(const v4f*)&B2[ub + hi8 + 4]);
      acc = wmma16(a0, b0, acc);
      acc = wmma16(a1, b1, acc);
      act_store(&ACTA[l16*1032 + ub + hi8], acc);
    }
    __syncthreads();

    // ---------------- Stage D: 16 x (64->1) + sigmoid, one lane per (row,o)
    {
      const int row = tid >> 4, o = tid & 15;
      const _Float16* h = (o == 0) ? &ACTA[row*1032] : &ACTB[row*1032 + o*64];
      const float*    w = &W3[o*64];
      float s = B3[o];
      #pragma unroll
      for (int c = 0; c < 8; c++) {
        v8h hh = *(const v8h*)&h[c*8];
        v4f w0 = *(const v4f*)&w[c*8];
        v4f w1 = *(const v4f*)&w[c*8 + 4];
        s += (float)hh[0]*w0[0] + (float)hh[1]*w0[1] +
             (float)hh[2]*w0[2] + (float)hh[3]*w0[3] +
             (float)hh[4]*w1[0] + (float)hh[5]*w1[1] +
             (float)hh[6]*w1[2] + (float)hh[7]*w1[3];
      }
      s = 1.f / (1.f + __expf(-s));
      const int grow = rowbase + row;
      if (grow < Bsz) out[(size_t)grow*16 + o] = s;
    }
    __syncthreads();  // protect ACTA/ACTB before next tile overwrites
  }
}

extern "C" void kernel_launch(void* const* d_in, const int* in_sizes, int n_in,
                              void* d_out, int out_size, void* d_ws, size_t ws_size,
                              hipStream_t stream) {
  (void)n_in; (void)out_size; (void)d_ws; (void)ws_size;
  const float* x     = (const float*)d_in[0];
  const float* dg_w0 = (const float*)d_in[1];
  const float* dg_b0 = (const float*)d_in[2];
  const float* dg_w1 = (const float*)d_in[3];
  const float* dg_b1 = (const float*)d_in[4];
  const float* dg_w2 = (const float*)d_in[5];
  const float* dg_b2 = (const float*)d_in[6];
  const float* dg_w3 = (const float*)d_in[7];
  const float* dg_b3 = (const float*)d_in[8];
  const float* ev_w0 = (const float*)d_in[9];
  const float* ev_b0 = (const float*)d_in[10];
  const float* ev_w1 = (const float*)d_in[11];
  const float* ev_b1 = (const float*)d_in[12];
  const float* ev_w2 = (const float*)d_in[13];
  const float* ev_b2 = (const float*)d_in[14];
  const float* ac_w0 = (const float*)d_in[15];
  const float* ac_b0 = (const float*)d_in[16];
  const float* ac_w1 = (const float*)d_in[17];
  const float* ac_b1 = (const float*)d_in[18];
  const float* ac_w2 = (const float*)d_in[19];
  const float* ac_b2 = (const float*)d_in[20];

  const int Bsz = in_sizes[0] / 20;
  const int ntiles = Bsz >> 4;
  int grid = ntiles < 1024 ? (ntiles > 0 ? ntiles : 1) : 1024;

  (void)hipFuncSetAttribute((const void*)actor_fused_kernel,
                            hipFuncAttributeMaxDynamicSharedMemorySize,
                            (int)SMEM_BYTES);

  actor_fused_kernel<<<grid, NTHREADS, SMEM_BYTES, stream>>>(
      x, dg_w0, dg_b0, dg_w1, dg_b1, dg_w2, dg_b2, dg_w3, dg_b3,
      ev_w0, ev_b0, ev_w1, ev_b1, ev_w2, ev_b2,
      ac_w0, ac_b0, ac_w1, ac_b1, ac_w2, ac_b2,
      (float*)d_out, Bsz);
}
